// TNetwork_76347338654230
// MI455X (gfx1250) — matrix operand order repaired and checked by reading, not compiled, not run on hardware
//
#include <hip/hip_runtime.h>
#include <math.h>

typedef __attribute__((ext_vector_type(16))) _Float16 v16h;
typedef __attribute__((ext_vector_type(8)))  float    v8f;

#define Bsz 256
#define Tsz 1024
#define Vsz 128
#define Hsz 128
#define Osz 2
#define MB  16        // batch rows per block (one WMMA M tile)
#define HSTRIDE 136   // halves per h-row in LDS (272B rows: 16B aligned, bank-optimal)
#define SCH 128       // symbol staging chunk (timesteps)

// Branch-free tanh: native v_tanh_f32 (confirmed in round-2 asm).
__device__ __forceinline__ float fast_tanh(float x) {
#if __has_builtin(__builtin_amdgcn_tanhf)
  return __builtin_amdgcn_tanhf(x);
#else
  const float e = __expf(2.0f * x);
  return 1.0f - 2.0f * __builtin_amdgcn_rcpf(e + 1.0f);
#endif
}

// ---------------- kernel 1: one-hot (B,T,V) f32 -> symbol id ----------------
// x is exactly one-hot, so sym = sum_v v * x[v]. One wave per (b,t);
// 134 MB of coalesced float4 reads -> pure HBM-bandwidth kernel (~6 us).
__global__ void k_symbols(const float* __restrict__ x, int* __restrict__ sym) {
  const int wave = (blockIdx.x * blockDim.x + threadIdx.x) >> 5;
  const int lane = threadIdx.x & 31;
  const float4 v = *(const float4*)(x + (size_t)wave * Vsz + lane * 4);
  float acc = v.x * (float)(lane * 4 + 0) + v.y * (float)(lane * 4 + 1)
            + v.z * (float)(lane * 4 + 2) + v.w * (float)(lane * 4 + 3);
  #pragma unroll
  for (int off = 16; off > 0; off >>= 1) acc += __shfl_xor(acc, off, 32);
  if (lane == 0) sym[wave] = (int)(acc + 0.5f);
}

// ---------------- kernel 2: recurrence via v_wmma_f32_16x16x32_f16 ----------
__global__ __launch_bounds__(256, 1)
void k_rnn(const int* __restrict__ sym, const float* __restrict__ W_ih,
           const float* __restrict__ b_ih, const float* __restrict__ b_hh,
           const float* __restrict__ W_hh, const float* __restrict__ W_out,
           const float* __restrict__ b_out, float* __restrict__ out) {
  __shared__ float s_Wp[Vsz * Hsz];                    // 64 KB gather table
  __shared__ __align__(16) _Float16 s_h0[MB * HSTRIDE];
  __shared__ __align__(16) _Float16 s_h1[MB * HSTRIDE];
  __shared__ __align__(16) int s_sym[SCH][MB];         // [tt][m]: lane's 8 syms contiguous

  const int tid   = threadIdx.x;
  const int lane  = tid & 31;
  const int w     = tid >> 5;          // wave id == N tile (8 x 16 = H)
  const int hi    = (lane >> 4) & 1;
  const int l15   = lane & 15;
  const int nglob = w * 16 + l15;      // column owned by this lane in B/C/D layout
  const int b0    = blockIdx.x * MB;
  const int arow  = l15 * HSTRIDE;     // A: M = lane%16

  // Build Wp[s][h] = W_ih[h][s-1] + b_ih[h] + b_hh[h] (Wp[0][*]=0) in LDS.
  for (int i = tid; i < Vsz * Hsz; i += 256) {
    const int s = i >> 7, h = i & 127;
    s_Wp[i] = (s > 0) ? (W_ih[h * (Vsz - 1) + (s - 1)] + b_ih[h] + b_hh[h]) : 0.0f;
  }
  for (int i = tid; i < MB * HSTRIDE; i += 256) s_h0[i] = (_Float16)0.0f;

  // Preload B = W_hh^T fragments, f32 -> f16 once. 32x16 f16 B layout (matches A):
  // lanes 0-15 hold K=0..7,16..23, lanes 16-31 hold K=8..15,24..31; N = lane%16.
  v16h bfr[4];
  #pragma unroll
  for (int kc = 0; kc < 4; ++kc) {
    const int kbase = kc * 32 + (hi ? 8 : 0);
    const float* wrow = W_hh + (size_t)nglob * Hsz;    // B[k][n] = W_hh[n][k]
    v16h bv;
    #pragma unroll
    for (int j = 0; j < 16; ++j) bv[j] = (_Float16)wrow[kbase + (j < 8 ? j : j + 8)];
    bfr[kc] = bv;
  }
  __syncthreads();

  float hreg[8];                       // this lane's h values (rows hi*8+r, col nglob)
  #pragma unroll
  for (int r = 0; r < 8; ++r) hreg[r] = 0.0f;

  // One timestep: read hsrc, write hdst; h state also carried in hreg.
  auto step = [&](const _Float16* __restrict__ hsrc, _Float16* __restrict__ hdst,
                  int tt) {
    const int* sp = &s_sym[tt][hi * 8];                // 2 x ds_load_b128
    const int4 sA = *(const int4*)(sp);
    const int4 sB = *(const int4*)(sp + 4);
    const int sv[8] = { sA.x, sA.y, sA.z, sA.w, sB.x, sB.y, sB.z, sB.w };

    // 4 fully independent WMMAs (literal-0 C), merged by a pk_add tree.
    v8f c[4];
    #pragma unroll
    for (int kc = 0; kc < 4; ++kc) {
      union { v16h v; uint4 q[2]; } a;
      const int kb = kc * 32 + (hi ? 8 : 0);           // 16-bit A 16x32 layout
      a.q[0] = *(const uint4*)(hsrc + arow + kb);      // halves K = kb .. kb+7
      a.q[1] = *(const uint4*)(hsrc + arow + kb + 16); // halves K = kb+16 .. kb+23
      v8f z;
      #pragma unroll
      for (int r = 0; r < 8; ++r) z[r] = 0.0f;
      c[kc] = __builtin_amdgcn_wmma_f32_16x16x32_f16(false, a.v, false, bfr[kc],
                                                     (short)0, z, false, false);
    }

    // epilogue: h = active ? tanh(acc + Wp[sym]) : h   (C/D: M=r+8*hi, N=lane%16)
    #pragma unroll
    for (int r = 0; r < 8; ++r) {
      const int s = sv[r];
      const float pre = s_Wp[s * Hsz + nglob];         // Wp[0][*] == 0
      const float acc = (c[0][r] + c[1][r]) + (c[2][r] + c[3][r]);
      const float hn = fast_tanh(acc + pre);
      const float res = s ? hn : hreg[r];
      hreg[r] = res;
      hdst[(r + hi * 8) * HSTRIDE + nglob] = (_Float16)res;
    }
    __syncthreads();
  };

  for (int t0 = 0; t0 < Tsz; t0 += SCH) {
    {  // stage SCH timesteps of symbols, transposed to [tt][m]
      const int m = tid >> 4, i0 = (tid & 15) * (SCH / 16);
      #pragma unroll
      for (int q = 0; q < SCH / 16; ++q)
        s_sym[i0 + q][m] = sym[(size_t)(b0 + m) * Tsz + t0 + i0 + q];
    }
    // pull next chunk's symbols toward the WGP (global_prefetch_b8)
    if (t0 + SCH < Tsz && tid < MB)
      __builtin_prefetch(&sym[(size_t)(b0 + tid) * Tsz + t0 + SCH], 0, 3);
    __syncthreads();
    for (int tt = 0; tt < SCH; tt += 2) {  // explicit ping-pong: reg+imm LDS addrs
      step(s_h0, s_h1, tt);
      step(s_h1, s_h0, tt + 1);
    }
  }
  // T even => final h in s_h0

  // hidden output: d_out[B*O + (b0+m)*H + k]
  for (int i = tid; i < MB * Hsz; i += 256) {
    const int m = i >> 7, k = i & 127;
    out[Bsz * Osz + (size_t)(b0 + m) * Hsz + k] = (float)s_h0[m * HSTRIDE + k];
  }
  // fed = softmax(h_last @ W_out^T + b_out), O = 2
  if (tid < MB) {
    const int m = tid;
    float z0 = b_out[0], z1 = b_out[1];
    for (int k = 0; k < Hsz; ++k) {
      const float hv = (float)s_h0[m * HSTRIDE + k];
      z0 += hv * W_out[k];
      z1 += hv * W_out[Hsz + k];
    }
    const float zm = fmaxf(z0, z1);
    const float e0 = __expf(z0 - zm), e1 = __expf(z1 - zm);
    const float inv = 1.0f / (e0 + e1);
    out[(b0 + m) * Osz + 0] = e0 * inv;
    out[(b0 + m) * Osz + 1] = e1 * inv;
  }
}

extern "C" void kernel_launch(void* const* d_in, const int* in_sizes, int n_in,
                              void* d_out, int out_size, void* d_ws, size_t ws_size,
                              hipStream_t stream) {
  (void)in_sizes; (void)n_in; (void)out_size; (void)ws_size;
  const float* x     = (const float*)d_in[0];
  const float* W_ih  = (const float*)d_in[1];
  const float* W_hh  = (const float*)d_in[2];
  const float* b_ih  = (const float*)d_in[3];
  const float* b_hh  = (const float*)d_in[4];
  const float* W_out = (const float*)d_in[5];
  const float* b_out = (const float*)d_in[6];
  float* out = (float*)d_out;

  int* sym = (int*)d_ws;                                   // 256*1024 ints = 1 MB

  k_symbols<<<(Bsz * Tsz) / 8, 256, 0, stream>>>(x, sym);  // one wave per (b,t)
  k_rnn<<<Bsz / MB, 256, 0, stream>>>(sym, W_ih, b_ih, b_hh, W_hh, W_out, b_out, out);
}
